// GNNPolicy_50036368999169
// MI455X (gfx1250) — matrix operand order repaired and checked by reading, not compiled, not run on hardware
//
#include <hip/hip_runtime.h>
#include <hip/hip_bf16.h>

// ---------------- CDNA5 WMMA types ----------------
typedef __attribute__((ext_vector_type(16))) __bf16       v16bf;
typedef __attribute__((ext_vector_type(8)))  float        v8f;
typedef __attribute__((ext_vector_type(8)))  unsigned int v8u;

// ---------------- helpers ----------------
__device__ __forceinline__ unsigned short f2bf(float f) {
    unsigned int u = __float_as_uint(f);
    unsigned int r = u + 0x7FFFu + ((u >> 16) & 1u);   // RNE
    return (unsigned short)(r >> 16);
}
// order-preserving float<->uint for atomicMax
__device__ __forceinline__ unsigned int fenc(float f) {
    unsigned int u = __float_as_uint(f);
    return (u & 0x80000000u) ? ~u : (u | 0x80000000u);
}
__device__ __forceinline__ float fdec(unsigned int e) {
    unsigned int u = (e & 0x80000000u) ? (e ^ 0x80000000u) : ~e;
    return __uint_as_float(u);
}
#define FENC_NEG_INF 0x007FFFFFu   // fenc(-inf)

// ---------------- generic fill ----------------
__global__ void k_fill_u32(unsigned int* p, unsigned int v, long long n) {
    long long i = (long long)blockIdx.x * blockDim.x + threadIdx.x;
    if (i < n) p[i] = v;
}

// ---------------- pack weight [K,Nc] f32 -> WMMA bf16 B-fragment layout ----------------
// idx = ((tn*kt + tk)*32 + lane)*16 + j ; k = tk*32 + (j>>3)*16 + (lane>>4)*8 + (j&7) ; n = tn*16 + (lane&15)
__global__ void k_pack_b(const float* __restrict__ W, unsigned short* __restrict__ Wpk,
                         int K, int Nc) {
    int idx = blockIdx.x * blockDim.x + threadIdx.x;
    int kt = K >> 5;
    int total = (Nc >> 4) * kt * 512;
    if (idx >= total) return;
    int j    = idx & 15;
    int lane = (idx >> 4) & 31;
    int blk  = idx >> 9;
    int tk   = blk % kt;
    int tn   = blk / kt;
    int k = tk * 32 + ((j >> 3) << 4) + ((lane >> 4) << 3) + (j & 7);
    int n = tn * 16 + (lane & 15);
    Wpk[idx] = f2bf(W[(size_t)k * Nc + n]);
}

// ---------------- MLP layer1: h1 = relu(x@W1 + b1), K=5, out bf16 [N,128] ----------------
__global__ void k_mlp1(const float* __restrict__ x, const float* __restrict__ W1,
                       const float* __restrict__ b1, unsigned short* __restrict__ h1bf, int N) {
    int idx = blockIdx.x * blockDim.x + threadIdx.x;
    if (idx >= N * 128) return;
    int n = idx >> 7, j = idx & 127;
    const float* xr = x + (size_t)n * 5;
    float acc = b1[j];
#pragma unroll
    for (int k = 0; k < 5; ++k) acc += xr[k] * W1[k * 128 + j];
    acc = acc > 0.f ? acc : 0.f;
    h1bf[idx] = f2bf(acc);
}

// ---------------- WMMA bf16 GEMM: C[M,Nc] = A[M,K](bf16) @ Wpk + bias ----------------
// block = 128 threads = 4 waves; wave w -> n-tile (blockIdx.y*4+w); blockIdx.x -> m-tile
__global__ __launch_bounds__(128) void k_gemm_wmma(
    const unsigned short* __restrict__ A, const unsigned short* __restrict__ Wpk,
    const float* __restrict__ bias, float* __restrict__ Cf,
    unsigned short* __restrict__ Cbf, int M, int K, int Nc, int relu) {
    int wave = threadIdx.x >> 5, lane = threadIdx.x & 31;
    int tm = blockIdx.x;
    int tn = blockIdx.y * 4 + wave;
    int half = lane >> 4;
    int kt = K >> 5;
    int m = tm * 16 + (lane & 15);
    int mload = m < M ? m : (M - 1);
    const unsigned short* arow  = A + (size_t)mload * K + half * 8;
    const unsigned short* bbase = Wpk + (((size_t)tn * kt) * 32 + lane) * 16;
    v8f acc = {};
    for (int t = 0; t < kt; ++t) {
        uint4 a0 = *(const uint4*)(arow);        // K = half*8 + 0..7
        uint4 a1 = *(const uint4*)(arow + 16);   // K = 16 + half*8 + 0..7
        uint4 b0 = *(const uint4*)(bbase);
        uint4 b1 = *(const uint4*)(bbase + 8);
        v8u au = {a0.x, a0.y, a0.z, a0.w, a1.x, a1.y, a1.z, a1.w};
        v8u bu = {b0.x, b0.y, b0.z, b0.w, b1.x, b1.y, b1.z, b1.w};
        acc = __builtin_amdgcn_wmma_f32_16x16x32_bf16(
            false, __builtin_bit_cast(v16bf, au),
            false, __builtin_bit_cast(v16bf, bu),
            (short)0, acc, false, false);
        arow += 32;
        bbase += 512;
    }
    int n = tn * 16 + (lane & 15);
    float bv = bias ? bias[n] : 0.f;
    int mbase = tm * 16 + half * 8;   // VGPR r: M = r (lanes 0-15) / r+8 (lanes 16-31)
#pragma unroll
    for (int r = 0; r < 8; ++r) {
        int mr = mbase + r;
        if (mr >= M) break;
        float v = acc[r] + bv;
        if (relu) v = v > 0.f ? v : 0.f;
        size_t o = (size_t)mr * Nc + n;
        if (Cf)  Cf[o] = v;
        if (Cbf) Cbf[o] = f2bf(v);
    }
}

// ---------------- edge pass A: logits + segment max ----------------
__global__ void k_edge_logits(const int* __restrict__ src, const int* __restrict__ dst,
                              int E0, int Etot, const float* __restrict__ xl,
                              const float* __restrict__ xr, const float* __restrict__ att,
                              int H, int D, float* __restrict__ logits,
                              unsigned int* __restrict__ mEnc) {
    int gw = (int)(((long long)blockIdx.x * blockDim.x + threadIdx.x) >> 5);
    int lane = threadIdx.x & 31;
    if (gw >= Etot * H) return;
    int e = gw / H, h = gw - e * H;
    int s, d; bool keep;
    if (e < E0) { s = src[e]; d = dst[e]; keep = (s != d); }
    else        { s = d = e - E0; keep = true; }
    const float* pl = xl + (size_t)s * H * D + (size_t)h * D;
    const float* pr = xr + (size_t)d * H * D + (size_t)h * D;
    const float* pa = att + (size_t)h * D;
    float acc = 0.f;
    for (int dd = lane; dd < D; dd += 32) {
        float t = pl[dd] + pr[dd];
        t = t > 0.f ? t : 0.2f * t;     // leaky relu
        acc += t * pa[dd];
    }
#pragma unroll
    for (int off = 16; off; off >>= 1) acc += __shfl_xor(acc, off, 32);
    if (lane == 0) {
        float lg = keep ? acc : -INFINITY;
        logits[(size_t)e * H + h] = lg;
        if (keep) atomicMax(&mEnc[(size_t)d * H + h], fenc(lg));
    }
}

// ---------------- edge pass B: exp + segment sum ----------------
__global__ void k_edge_exp(const int* __restrict__ src, const int* __restrict__ dst,
                           int E0, int Etot, const float* __restrict__ logits,
                           const unsigned int* __restrict__ mEnc,
                           float* __restrict__ exbuf, float* __restrict__ denom, int H) {
    int idx = blockIdx.x * blockDim.x + threadIdx.x;
    if (idx >= Etot * H) return;
    int e = idx / H, h = idx - e * H;
    int d; bool keep;
    if (e < E0) { int s = src[e]; d = dst[e]; keep = (s != d); }
    else        { d = e - E0; keep = true; }
    float ex = 0.f;
    if (keep) {
        float m = fdec(mEnc[(size_t)d * H + h]);
        ex = __expf(logits[idx] - m);
    }
    exbuf[idx] = ex;
    if (ex > 0.f) atomicAdd(&denom[(size_t)d * H + h], ex);
}

// ---------------- edge pass C: out[dst] += alpha * xl[src] ----------------
__global__ void k_edge_scatter(const int* __restrict__ src, const int* __restrict__ dst,
                               int E0, long long Etot, const float* __restrict__ exbuf,
                               const float* __restrict__ denom, const float* __restrict__ xl,
                               float* __restrict__ outb, int H, int D) {
    long long idx = (long long)blockIdx.x * blockDim.x + threadIdx.x;
    int hd = H * D;
    if (idx >= Etot * hd) return;
    int e = (int)(idx / hd);
    int c = (int)(idx - (long long)e * hd);
    int h = c / D;
    int s, d;
    if (e < E0) { s = src[e]; d = dst[e]; }
    else        { s = d = e - E0; }
    float ex = exbuf[(size_t)e * H + h];
    if (ex == 0.f) return;
    float alpha = ex / (denom[(size_t)d * H + h] + 1e-16f);
    atomicAdd(&outb[(size_t)d * hd + c], alpha * xl[(size_t)s * hd + c]);
}

// ---------------- finish gnn1: g1 = bf16(relu(scat + bias1)) ----------------
__global__ void k_finish1(const float* __restrict__ scat, const float* __restrict__ bias,
                          unsigned short* __restrict__ g1bf, long long total, int C) {
    long long idx = (long long)blockIdx.x * blockDim.x + threadIdx.x;
    if (idx >= total) return;
    int c = (int)(idx % C);
    float v = scat[idx] + bias[c];
    v = v > 0.f ? v : 0.f;
    g1bf[idx] = f2bf(v);
}

// ---------------- final bias add (in place on d_out) ----------------
__global__ void k_bias_out(float* __restrict__ out, const float* __restrict__ bias,
                           long long total, int C) {
    long long idx = (long long)blockIdx.x * blockDim.x + threadIdx.x;
    if (idx >= total) return;
    out[idx] += bias[(int)(idx % C)];
}

// ======================================================================
extern "C" void kernel_launch(void* const* d_in, const int* in_sizes, int n_in,
                              void* d_out, int out_size, void* d_ws, size_t ws_size,
                              hipStream_t stream) {
    const float* x     = (const float*)d_in[0];
    const int*   eidx  = (const int*)d_in[1];
    const float* W1    = (const float*)d_in[2];
    const float* b1    = (const float*)d_in[3];
    const float* W2    = (const float*)d_in[4];
    const float* b2    = (const float*)d_in[5];
    const float* Wl1   = (const float*)d_in[6];
    const float* bl1   = (const float*)d_in[7];
    const float* Wr1   = (const float*)d_in[8];
    const float* br1   = (const float*)d_in[9];
    const float* att1  = (const float*)d_in[10];
    const float* bias1 = (const float*)d_in[11];
    const float* Wl2   = (const float*)d_in[12];
    const float* bl2   = (const float*)d_in[13];
    const float* Wr2   = (const float*)d_in[14];
    const float* br2   = (const float*)d_in[15];
    const float* att2  = (const float*)d_in[16];
    const float* bias2 = (const float*)d_in[17];

    const int N    = in_sizes[0] / 5;
    const int E0   = in_sizes[1] / 2;
    const int Etot = E0 + N;
    const int* src = eidx;
    const int* dst = eidx + E0;
    const int Hd = 128;

    // ---- workspace bump allocator (256B aligned) ----
    char* ws = (char*)d_ws;
    size_t off = 0;
    auto alloc = [&](size_t bytes) -> void* {
        off = (off + 255) & ~(size_t)255;
        void* p = ws + off;
        off += bytes;
        return p;
    };
    unsigned short* h1bf  = (unsigned short*)alloc((size_t)N * 128 * 2);
    unsigned short* h0bf  = (unsigned short*)alloc((size_t)N * 128 * 2);
    float*          xl1   = (float*)alloc((size_t)N * 512 * 4);
    float*          xr1   = (float*)alloc((size_t)N * 512 * 4);
    float*          scat1 = (float*)alloc((size_t)N * 512 * 4);
    unsigned short* g1bf  = (unsigned short*)alloc((size_t)N * 512 * 2);
    float*          xl2   = (float*)alloc((size_t)N * 128 * 4);
    float*          xr2   = (float*)alloc((size_t)N * 128 * 4);
    float*          logit = (float*)alloc((size_t)Etot * 4 * 4);
    float*          exbuf = (float*)alloc((size_t)Etot * 4 * 4);
    unsigned int*   mEnc  = (unsigned int*)alloc((size_t)N * 4 * 4);
    float*          denom = (float*)alloc((size_t)N * 4 * 4);
    unsigned short* W2pk  = (unsigned short*)alloc((size_t)128 * 128 * 2);
    unsigned short* Wl1pk = (unsigned short*)alloc((size_t)128 * 512 * 2);
    unsigned short* Wr1pk = (unsigned short*)alloc((size_t)128 * 512 * 2);
    unsigned short* Wl2pk = (unsigned short*)alloc((size_t)512 * 128 * 2);
    unsigned short* Wr2pk = (unsigned short*)alloc((size_t)512 * 128 * 2);

    const int BS = 256;
    auto blk1 = [&](long long n) { return dim3((unsigned)((n + BS - 1) / BS)); };

    // ---- pack weights into WMMA B-fragment layout ----
    k_pack_b<<<blk1((128 / 16) * (128 / 32) * 512), BS, 0, stream>>>(W2,  W2pk,  128, 128);
    k_pack_b<<<blk1((512 / 16) * (128 / 32) * 512), BS, 0, stream>>>(Wl1, Wl1pk, 128, 512);
    k_pack_b<<<blk1((512 / 16) * (128 / 32) * 512), BS, 0, stream>>>(Wr1, Wr1pk, 128, 512);
    k_pack_b<<<blk1((128 / 16) * (512 / 32) * 512), BS, 0, stream>>>(Wl2, Wl2pk, 512, 128);
    k_pack_b<<<blk1((128 / 16) * (512 / 32) * 512), BS, 0, stream>>>(Wr2, Wr2pk, 512, 128);

    // ---- node embed MLP ----
    k_mlp1<<<blk1((long long)N * 128), BS, 0, stream>>>(x, W1, b1, h1bf, N);
    const int mt = (N + 15) / 16;
    // h0 = h1 @ W2 + b2  (bf16 out only)
    k_gemm_wmma<<<dim3(mt, 128 / 64), 128, 0, stream>>>(h1bf, W2pk, b2, nullptr, h0bf,
                                                        N, 128, 128, 0);
    // ---- gnn1 transforms ----
    k_gemm_wmma<<<dim3(mt, 512 / 64), 128, 0, stream>>>(h0bf, Wl1pk, bl1, xl1, nullptr,
                                                        N, 128, 512, 0);
    k_gemm_wmma<<<dim3(mt, 512 / 64), 128, 0, stream>>>(h0bf, Wr1pk, br1, xr1, nullptr,
                                                        N, 128, 512, 0);
    // ---- gnn1 edge phase (H=4, D=128) ----
    k_fill_u32<<<blk1((long long)N * 4), BS, 0, stream>>>(mEnc, FENC_NEG_INF, (long long)N * 4);
    k_fill_u32<<<blk1((long long)N * 4), BS, 0, stream>>>((unsigned int*)denom, 0u, (long long)N * 4);
    k_fill_u32<<<blk1((long long)N * 512), BS, 0, stream>>>((unsigned int*)scat1, 0u, (long long)N * 512);
    k_edge_logits<<<blk1((long long)Etot * 4 * 32), BS, 0, stream>>>(
        src, dst, E0, Etot, xl1, xr1, att1, 4, Hd, logit, mEnc);
    k_edge_exp<<<blk1((long long)Etot * 4), BS, 0, stream>>>(
        src, dst, E0, Etot, logit, mEnc, exbuf, denom, 4);
    k_edge_scatter<<<blk1((long long)Etot * 512), BS, 0, stream>>>(
        src, dst, E0, Etot, exbuf, denom, xl1, scat1, 4, Hd);
    k_finish1<<<blk1((long long)N * 512), BS, 0, stream>>>(scat1, bias1, g1bf,
                                                           (long long)N * 512, 512);
    // ---- gnn2 transforms (K=512 -> 128) ----
    k_gemm_wmma<<<dim3(mt, 128 / 64), 128, 0, stream>>>(g1bf, Wl2pk, bl2, xl2, nullptr,
                                                        N, 512, 128, 0);
    k_gemm_wmma<<<dim3(mt, 128 / 64), 128, 0, stream>>>(g1bf, Wr2pk, br2, xr2, nullptr,
                                                        N, 512, 128, 0);
    // ---- gnn2 edge phase (H=1, D=128), scatter directly into d_out ----
    float* out = (float*)d_out;
    k_fill_u32<<<blk1((long long)N), BS, 0, stream>>>(mEnc, FENC_NEG_INF, (long long)N);
    k_fill_u32<<<blk1((long long)N), BS, 0, stream>>>((unsigned int*)denom, 0u, (long long)N);
    k_fill_u32<<<blk1((long long)N * 128), BS, 0, stream>>>((unsigned int*)out, 0u, (long long)N * 128);
    k_edge_logits<<<blk1((long long)Etot * 32), BS, 0, stream>>>(
        src, dst, E0, Etot, xl2, xr2, att2, 1, Hd, logit, mEnc);
    k_edge_exp<<<blk1((long long)Etot), BS, 0, stream>>>(
        src, dst, E0, Etot, logit, mEnc, exbuf, denom, 1);
    k_edge_scatter<<<blk1((long long)Etot * 128), BS, 0, stream>>>(
        src, dst, E0, Etot, exbuf, denom, xl2, out, 1, Hd);
    k_bias_out<<<blk1((long long)N * 128), BS, 0, stream>>>(out, bias2, (long long)N * 128, 128);
    (void)n_in; (void)out_size; (void)ws_size;
}
